// SelfAttention_33535104647986
// MI455X (gfx1250) — matrix operand order repaired and checked by reading, not compiled, not run on hardware
//
#include <hip/hip_runtime.h>

typedef unsigned short u16;
typedef __attribute__((ext_vector_type(8)))  unsigned short u16x8;
typedef __attribute__((ext_vector_type(16))) unsigned short u16x16;
typedef __attribute__((ext_vector_type(16))) __bf16        v16bf;
typedef __attribute__((ext_vector_type(8)))  float         v8f;

// ---------- helpers ----------
static __device__ __forceinline__ u16 f2bf(float f) {
    unsigned u = __builtin_bit_cast(unsigned, f);
    u += 0x7FFFu + ((u >> 16) & 1u);          // round-to-nearest-even
    return (u16)(u >> 16);
}

// async global->LDS copy of 16 bytes per lane (CDNA5 async DMA, ASYNCcnt)
static __device__ __forceinline__ void async_copy_b128(void* lds, const void* gsrc) {
    unsigned ldsaddr = (unsigned)(unsigned long long)lds;   // low 32 bits = LDS offset
    asm volatile("global_load_async_to_lds_b128 %0, %1, off"
                 :: "v"(ldsaddr), "v"(gsrc)
                 : "memory");
}
// counted wait: proceed when <= N async ops outstanding (in-order completion)
template <int N>
static __device__ __forceinline__ void wait_async() {
    asm volatile("s_wait_asynccnt %0" :: "i"(N) : "memory");
}

// raw max (no canonicalize)
static __device__ __forceinline__ float fmax_raw(float a, float b) {
    float d;
    asm("v_max_num_f32 %0, %1, %2" : "=v"(d) : "v"(a), "v"(b));
    return d;
}
// lane permute within each 16-lane row, distinct dst (no old-operand copy)
template <unsigned S0, unsigned S1>
static __device__ __forceinline__ float pl16(float v) {
    float d;
    asm("v_permlane16_b32 %0, %1, %2, %3" : "=v"(d) : "v"(v), "s"(S0), "s"(S1));
    return d;
}
static __device__ __forceinline__ float rowmax16(float v) {
    v = fmax_raw(v, pl16<0x67452301u, 0xEFCDAB89u>(v));   // xor 1
    v = fmax_raw(v, pl16<0x54761032u, 0xDCFE98BAu>(v));   // xor 2
    v = fmax_raw(v, pl16<0x32107654u, 0xBA98FEDCu>(v));   // xor 4
    v = fmax_raw(v, pl16<0xFEDCBA98u, 0x76543210u>(v));   // xor 8
    return v;
}
static __device__ __forceinline__ float rowbcast0(float v) {
    return pl16<0x00000000u, 0x00000000u>(v);
}

// fragment from two contiguous 8x16-bit LDS runs (A-frags, row-contiguous B-frags)
static __device__ __forceinline__ v16bf frag2x8(const u16* lo_p, const u16* hi_p) {
    u16x8 lo = *reinterpret_cast<const u16x8*>(lo_p);
    u16x8 hi = *reinterpret_cast<const u16x8*>(hi_p);
    u16x16 f;
#pragma unroll
    for (int i = 0; i < 8; ++i) { f[i] = lo[i]; f[i + 8] = hi[i]; }
    return __builtin_bit_cast(v16bf, f);
}
static __device__ __forceinline__ v16bf frag16(const u16* p) {
    return frag2x8(p, p + 8);
}
// B-fragment by strided gather: element e = p[e*stride]  (k contiguous down column)
static __device__ __forceinline__ v16bf fragB_strided(const u16* p, int stride) {
    u16x16 f;
#pragma unroll
    for (int i = 0; i < 16; ++i) f[i] = p[i * stride];
    return __builtin_bit_cast(v16bf, f);
}

// ---------- fp32 -> bf16 convert ----------
__global__ void f32_to_bf16_kernel(const float* __restrict__ in,
                                   u16* __restrict__ out, int n) {
    int i = blockIdx.x * blockDim.x + threadIdx.x;
    if (i < n) out[i] = f2bf(in[i]);
}

// ---------- generic bf16 GEMM: C = ((A*B) + bias) * oscale ----------
// A row-major bf16 (lda=K), B row-major bf16 (ldb=N).
// Tile: 128(M) x 128(N), K-step 32. 256 threads = 8 waves.
// Double-buffered async LDS-DMA staging, counted asynccnt waits.
template <bool F32OUT>
__global__ __launch_bounds__(256) void gemm_bf16_kernel(
    const u16* __restrict__ A, const u16* __restrict__ B,
    const float* __restrict__ bias,
    u16* __restrict__ Cb, float* __restrict__ Cf,
    int M, int N, int K, float oscale) {
    __shared__ __attribute__((aligned(16))) u16 sA0[128 * 32], sA1[128 * 32]; // [m][k]
    __shared__ __attribute__((aligned(16))) u16 sB0[32 * 128], sB1[32 * 128]; // [k][n]

    const int t = threadIdx.x;
    const int wave = t >> 5;
    const int lane = t & 31;
    const int half = lane >> 4;           // 0 or 1
    const int lid  = lane & 15;
    const int mw   = wave & 3;            // 0..3  -> 32-row strip
    const int nw   = wave >> 2;           // 0..1  -> 64-col strip
    const int m0 = blockIdx.y * 128;
    const int n0 = blockIdx.x * 128;

    const v8f vz = {0.f, 0.f, 0.f, 0.f, 0.f, 0.f, 0.f, 0.f};
    v8f acc[2][4] = {{vz, vz, vz, vz}, {vz, vz, vz, vz}};

    const int ar  = t >> 1;               // 0..127 A row
    const int ako = (t & 1) * 16;         // 0 / 16

    // stage tile for k-offset kk into (dA, dB): 4 async b128 per thread
    auto stage = [&](int kk, u16* dA, u16* dB) {
        const u16* src = A + (size_t)(m0 + ar) * K + kk + ako;
        async_copy_b128(&dA[ar * 32 + ako],     src);
        async_copy_b128(&dA[ar * 32 + ako + 8], src + 8);
#pragma unroll
        for (int j = 0; j < 2; ++j) {
            int c   = t + 256 * j;            // 0..511 chunks of 8 elems
            int kr  = c >> 4;                 // 0..31
            int off = (c & 15) * 8;           // 0..120
            async_copy_b128(&dB[kr * 128 + off],
                            B + (size_t)(kk + kr) * N + n0 + off);
        }
    };

    stage(0, sA0, sB0);
    int pb = 0;
    for (int kk = 0; kk < K; kk += 32, pb ^= 1) {
        const u16* cA = pb ? sA1 : sA0;
        const u16* cB = pb ? sB1 : sB0;
        if (kk + 32 < K) {
            stage(kk + 32, pb ? sA0 : sA1, pb ? sB0 : sB1);
            wait_async<4>();                  // current tile resident; next streams
        } else {
            wait_async<0>();
        }
        __syncthreads();

        // A fragments for this wave (rows mw*32 + {0,16} + lid)
        const int kb = half * 16;
        const u16* ap0 = &cA[(mw * 32 + lid) * 32];
        const u16* ap1 = &cA[(mw * 32 + 16 + lid) * 32];
        v16bf af0 = frag2x8(ap0 + kb, ap0 + kb + 8);
        v16bf af1 = frag2x8(ap1 + kb, ap1 + kb + 8);

#pragma unroll
        for (int ns = 0; ns < 4; ++ns) {
            // B-frag: column n fixed, k contiguous (element e -> k = e + 16*half)
            v16bf bf = fragB_strided(&cB[(half * 16) * 128 + nw * 64 + ns * 16 + lid], 128);
            acc[0][ns] = __builtin_amdgcn_wmma_f32_16x16x32_bf16(
                false, af0, false, bf, (short)0, acc[0][ns], false, false);
            acc[1][ns] = __builtin_amdgcn_wmma_f32_16x16x32_bf16(
                false, af1, false, bf, (short)0, acc[1][ns], false, false);
        }
        __syncthreads();   // reads done before this buffer is re-staged
    }

    // store C fragments: element r -> m = r + 8*half, n = lid
#pragma unroll
    for (int mf = 0; mf < 2; ++mf) {
#pragma unroll
        for (int ns = 0; ns < 4; ++ns) {
            const int gcol = n0 + nw * 64 + ns * 16 + lid;
            const float bcol = bias[gcol];
            const int mbase = m0 + mw * 32 + mf * 16 + half * 8;
#pragma unroll
            for (int r = 0; r < 8; ++r) {
                float vo = (acc[mf][ns][r] + bcol) * oscale;
                size_t idx = (size_t)(mbase + r) * N + gcol;
                if (F32OUT) Cf[idx] = vo;
                else        Cb[idx] = f2bf(vo);
            }
        }
    }
}

// ---------- flash attention ----------
// Q (pre-scaled by 1/sqrt(hd)), K, V: bf16 [B*L][512]; head h at cols h*64..h*64+63.
// grid = (L/64, B*H); block = 128 (4 waves); wave w owns q rows w*16..w*16+15.
// Double-buffered async K/V staging; row sums via ones-column WMMA; max via permlane.
__global__ __launch_bounds__(128) void flash_attn_kernel(
    const u16* __restrict__ Q, const u16* __restrict__ Kmat,
    const u16* __restrict__ V, u16* __restrict__ Y) {
    __shared__ __attribute__((aligned(16))) u16 sQ[64 * 64];                  // [q][d]
    __shared__ __attribute__((aligned(16))) u16 sK0[32 * 64], sK1[32 * 64];   // [key][d]
    __shared__ __attribute__((aligned(16))) u16 sV0[32 * 64], sV1[32 * 64];   // [key][d]
    __shared__ __attribute__((aligned(16))) u16 sP[4][16 * 32];               // [m][key]

    const int t    = threadIdx.x;
    const int wave = t >> 5;
    const int lane = t & 31;
    const int half = lane >> 4;
    const int lid  = lane & 15;

    const int q0 = blockIdx.x * 64;
    const int bh = blockIdx.y;
    const int b  = bh >> 3;
    const int h  = bh & 7;
    const size_t rowbase = (size_t)b * 2048;
    const int coff = h * 64;

    // stage K/V tile at key-offset key0: 4 async b128 per thread
    auto stage_kv = [&](int key0, u16* dK, u16* dV) {
#pragma unroll
        for (int j = 0; j < 2; ++j) {
            int c   = t + 128 * j;        // 0..255 chunks of 8 elems
            int r   = c >> 3;             // 0..31
            int off = (c & 7) * 8;        // 0..56
            const size_t grow = (rowbase + key0 + r) * 512 + coff + off;
            async_copy_b128(&dK[r * 64 + off], Kmat + grow);
            async_copy_b128(&dV[r * 64 + off], V + grow);
        }
    };

    // Q block (4 async b128 per thread), then prefetch first K/V tile
#pragma unroll
    for (int j = 0; j < 4; ++j) {
        int c   = t + 128 * j;            // 0..511 chunks of 8 elems
        int r   = c >> 3;                 // 0..63
        int off = (c & 7) * 8;            // 0..56
        async_copy_b128(&sQ[r * 64 + off],
                        Q + (rowbase + q0 + r) * 512 + coff + off);
    }
    stage_kv(0, sK0, sV0);
    wait_async<4>();                      // Q resident; tile 0 still streaming
    __syncthreads();

    // per-wave Q fragments (K-dim 64 -> two 32-chunks), register resident
    const int kb = half * 16;
    const u16* qp = &sQ[(wave * 16 + lid) * 64];
    v16bf qa0 = frag2x8(qp + 0  + kb, qp + 0  + kb + 8);
    v16bf qa1 = frag2x8(qp + 32 + kb, qp + 32 + kb + 8);

    // ones-column B fragment: B[k][n] = (n==0) ? 1.0 : 0.0 for all k
    v16bf vone;
    {
        u16 ov = (lid == 0) ? (u16)0x3F80 : (u16)0;
        u16x16 of;
#pragma unroll
        for (int i = 0; i < 16; ++i) of[i] = ov;
        vone = __builtin_bit_cast(v16bf, of);
    }

    const v8f vz = {0.f, 0.f, 0.f, 0.f, 0.f, 0.f, 0.f, 0.f};
    v8f o[5] = {vz, vz, vz, vz, vz};      // o[4] = softmax row-sum column
    float mrow[8];
#pragma unroll
    for (int r = 0; r < 8; ++r) mrow[r] = -3.0e38f;

    int pb = 0;
    for (int kt = 0; kt < 64; ++kt, pb ^= 1) {
        const u16* cK = pb ? sK1 : sK0;
        const u16* cV = pb ? sV1 : sV0;
        if (kt + 1 < 64) {
            stage_kv((kt + 1) * 32, pb ? sK0 : sK1, pb ? sV0 : sV1);
            wait_async<4>();              // tile kt resident; kt+1 streams
        } else {
            wait_async<0>();
        }
        __syncthreads();

        // S = Q * K^T for this wave: 16 q x 32 keys (two n-subtiles)
        v8f s0 = vz, s1 = vz;
        {
            const u16* kp0 = &cK[(0 * 16 + lid) * 64 + half * 16];
            const u16* kp1 = &cK[(1 * 16 + lid) * 64 + half * 16];
            v16bf b00 = frag16(kp0 + 0);
            v16bf b01 = frag16(kp0 + 32);
            v16bf b10 = frag16(kp1 + 0);
            v16bf b11 = frag16(kp1 + 32);
            s0 = __builtin_amdgcn_wmma_f32_16x16x32_bf16(false, qa0, false, b00, (short)0, s0, false, false);
            s0 = __builtin_amdgcn_wmma_f32_16x16x32_bf16(false, qa1, false, b01, (short)0, s0, false, false);
            s1 = __builtin_amdgcn_wmma_f32_16x16x32_bf16(false, qa0, false, b10, (short)0, s1, false, false);
            s1 = __builtin_amdgcn_wmma_f32_16x16x32_bf16(false, qa1, false, b11, (short)0, s1, false, false);
        }

        // online softmax per row (element r -> m = r + 8*half)
#pragma unroll
        for (int r = 0; r < 8; ++r) {
            float v0 = s0[r];
            float v1 = s1[r];
            float mx = rowmax16(fmax_raw(v0, v1));
            float mnew  = fmax_raw(mrow[r], mx);
            float alpha = __expf(mrow[r] - mnew);
            float p0 = __expf(v0 - mnew);
            float p1 = __expf(v1 - mnew);
            mrow[r] = mnew;
#pragma unroll
            for (int ns = 0; ns < 5; ++ns) o[ns][r] *= alpha;
            int m = r + half * 8;
            sP[wave][m * 32 + lid]      = f2bf(p0);
            sP[wave][m * 32 + 16 + lid] = f2bf(p1);
        }
        __syncthreads();

        // O += P * V  (contraction over 32 keys); o[4] += P * ones_col (row sums)
        {
            const u16* pp = &sP[wave][lid * 32];
            v16bf pa = frag2x8(pp + kb, pp + kb + 8);
#pragma unroll
            for (int ns = 0; ns < 4; ++ns) {
                // B-frag from [key][d]: column d fixed, k=key strided by 64
                v16bf bf = fragB_strided(&cV[(half * 16) * 64 + ns * 16 + lid], 64);
                o[ns] = __builtin_amdgcn_wmma_f32_16x16x32_bf16(
                    false, pa, false, bf, (short)0, o[ns], false, false);
            }
            o[4] = __builtin_amdgcn_wmma_f32_16x16x32_bf16(
                false, pa, false, vone, (short)0, o[4], false, false);
        }
        __syncthreads();   // reads done before this buffer is re-staged
    }

    // normalize and store Y (bf16); row sum sits in row-lane 0 of o[4]
#pragma unroll
    for (int r = 0; r < 8; ++r) {
        float rsum = rowbcast0(o[4][r]);
        float rinv = 1.0f / rsum;
        int m = r + half * 8;
        size_t row = rowbase + q0 + wave * 16 + m;
#pragma unroll
        for (int ns = 0; ns < 4; ++ns) {
            int col = coff + ns * 16 + lid;
            Y[row * 512 + col] = f2bf(o[ns][r] * rinv);
        }
    }
}

// ---------- launch ----------
extern "C" void kernel_launch(void* const* d_in, const int* in_sizes, int n_in,
                              void* d_out, int out_size, void* d_ws, size_t ws_size,
                              hipStream_t stream) {
    (void)in_sizes; (void)n_in; (void)out_size; (void)ws_size;
    const float* x  = (const float*)d_in[0];
    const float* Wq = (const float*)d_in[1];
    const float* bq = (const float*)d_in[2];
    const float* Wk = (const float*)d_in[3];
    const float* bk = (const float*)d_in[4];
    const float* Wv = (const float*)d_in[5];
    const float* bv = (const float*)d_in[6];
    const float* Wp = (const float*)d_in[7];
    const float* bp = (const float*)d_in[8];

    const int Btok = 4 * 2048;        // 8192 rows
    const int D    = 512;
    const int NX   = Btok * D;        // 4,194,304
    const int NW   = D * D;           // 262,144

    u16* ws  = (u16*)d_ws;
    u16* xb  = ws;
    u16* wqb = xb  + (size_t)NX;
    u16* wkb = wqb + NW;
    u16* wvb = wkb + NW;
    u16* wpb = wvb + NW;
    u16* Qb  = wpb + NW;
    u16* Kb  = Qb  + (size_t)NX;
    u16* Vb  = Kb  + (size_t)NX;
    u16* Yb  = Vb  + (size_t)NX;

    // convert fp32 -> bf16
    f32_to_bf16_kernel<<<(NX + 255) / 256, 256, 0, stream>>>(x,  xb,  NX);
    f32_to_bf16_kernel<<<(NW + 255) / 256, 256, 0, stream>>>(Wq, wqb, NW);
    f32_to_bf16_kernel<<<(NW + 255) / 256, 256, 0, stream>>>(Wk, wkb, NW);
    f32_to_bf16_kernel<<<(NW + 255) / 256, 256, 0, stream>>>(Wv, wvb, NW);
    f32_to_bf16_kernel<<<(NW + 255) / 256, 256, 0, stream>>>(Wp, wpb, NW);

    // QKV projections (bf16 out); Q pre-scaled by 1/sqrt(64)
    dim3 gg(D / 128, Btok / 128);     // (4, 64)
    gemm_bf16_kernel<false><<<gg, 256, 0, stream>>>(xb, wqb, bq, Qb, nullptr, Btok, D, D, 0.125f);
    gemm_bf16_kernel<false><<<gg, 256, 0, stream>>>(xb, wkb, bk, Kb, nullptr, Btok, D, D, 1.0f);
    gemm_bf16_kernel<false><<<gg, 256, 0, stream>>>(xb, wvb, bv, Vb, nullptr, Btok, D, D, 1.0f);

    // flash attention -> Y (bf16)
    flash_attn_kernel<<<dim3(2048 / 64, 4 * 8), 128, 0, stream>>>(Qb, Kb, Vb, Yb);

    // output projection (fp32 out + bias)
    gemm_bf16_kernel<true><<<gg, 256, 0, stream>>>(Yb, wpb, bp, nullptr, (float*)d_out, Btok, D, D, 1.0f);
}
